// MultiHeadAttention_1855425872096
// MI455X (gfx1250) — matrix operand order repaired and checked by reading, not compiled, not run on hardware
//
#include <hip/hip_runtime.h>
#include <hip/hip_bf16.h>

// Problem constants (match reference)
#define BB 2
#define SS 2048
#define DDIM 1024
#define HH 16
#define DHH 64
#define MM (BB * SS)   // 4096 rows for all projections

typedef __attribute__((ext_vector_type(16))) __bf16 v16bf;
typedef __attribute__((ext_vector_type(8)))  float  v8f;

union FragAB {
    v16bf v;
    unsigned int u[8];
    __bf16 h[16];
};

union Pack2 {
    __bf16 h[2];
    unsigned int u;
};

__device__ __forceinline__ v8f wmma_bf16(v16bf a, v16bf b, v8f c) {
    // (neg_a, A, neg_b, B, c_mod, C, reuse_a, reuse_b)
    return __builtin_amdgcn_wmma_f32_16x16x32_bf16(false, a, false, b, (short)0, c, false, false);
}

// ---------------------------------------------------------------------------
// Tiled bf16 GEMM:  Out = X[M,K] @ W[K,N] + bias
// Block tile 256x64, wave tile 32x64 (8 WMMAs per 32-k step, B-frags reused).
// MODE 0: bf16 out, head-split  [B,H,S,DH]   (for Q, K)
// MODE 1: f32  out, row-major   [M,N]        (final projection -> d_out)
// MODE 2: bf16 out, transposed  [B,H,DH,S]   (for V, so P@V B-frags are contiguous)
// XF32:   X is f32 (inputs) vs bf16 (attention output)
// ---------------------------------------------------------------------------
template<int MODE, bool XF32>
__global__ __launch_bounds__(256)
void gemm_kernel(const void* __restrict__ Xraw, const float* __restrict__ W,
                 const float* __restrict__ bias, void* __restrict__ Out)
{
    constexpr int K   = DDIM;
    constexpr int N   = DDIM;
    constexpr int BM  = 256, BN = 64, BK = 32;
    constexpr int AST = BK + 2;   // 34-element stride -> conflict-free frag reads
    constexpr int BST = BK + 2;

    __shared__ __bf16 As[BM * AST];   // [row][k]
    __shared__ __bf16 Bs[BN * BST];   // [n][k]  (W tile stored transposed)

    const int tid  = threadIdx.x;
    const int lane = tid & 31;
    const int wv   = tid >> 5;               // wave 0..7 -> 32-row slice
    const int m0   = blockIdx.x * BM;
    const int n0   = blockIdx.y * BN;

    const float*  Xf = (const float*) Xraw;
    const __bf16* Xb = (const __bf16*)Xraw;

    v8f acc[2][4];
    {
        v8f z = {};
        #pragma unroll
        for (int mh = 0; mh < 2; ++mh)
            #pragma unroll
            for (int nt = 0; nt < 4; ++nt) acc[mh][nt] = z;
    }

    for (int k0 = 0; k0 < K; k0 += BK) {
        // Stage A tile (256x32) as packed bf16 pairs. Coalesced 8B/lane.
        {
            const int p  = tid & 15;    // k-pair index
            const int r0 = tid >> 4;    // 16 rows per pass
            #pragma unroll
            for (int rr = 0; rr < BM / 16; ++rr) {
                const int row = r0 + rr * 16;
                Pack2 cv;
                if constexpr (XF32) {
                    const float* src = Xf + (long)(m0 + row) * K + k0 + 2 * p;
                    cv.h[0] = (__bf16)src[0];
                    cv.h[1] = (__bf16)src[1];
                } else {
                    cv.u = *(const unsigned int*)(Xb + (long)(m0 + row) * K + k0 + 2 * p);
                }
                *(unsigned int*)&As[row * AST + 2 * p] = cv.u;
            }
        }
        // Stage B tile (32x64) transposed -> Bs[n][k], packed pairs along k.
        {
            const int n   = tid & 63;
            const int kp0 = tid >> 6;   // 4 pairs per pass
            #pragma unroll
            for (int kk = 0; kk < 4; ++kk) {
                const int kp = kp0 + kk * 4;
                Pack2 cv;
                cv.h[0] = (__bf16)W[(long)(k0 + 2 * kp)     * N + n0 + n];
                cv.h[1] = (__bf16)W[(long)(k0 + 2 * kp + 1) * N + n0 + n];
                *(unsigned int*)&Bs[n * BST + 2 * kp] = cv.u;
            }
        }
        __syncthreads();

        // B fragments: loaded once, reused for both 16-row halves of the wave tile
        FragAB bfr[4];
        {
            const int kb = (lane < 16) ? 0 : 16;
            #pragma unroll
            for (int nt = 0; nt < 4; ++nt) {
                const int n = nt * 16 + (lane & 15);
                #pragma unroll
                for (int j = 0; j < 8; ++j)
                    bfr[nt].u[j] = *(const unsigned int*)&Bs[n * BST + ((j < 4) ? 0 : 16) + (kb >> 1) + 2 * (j & 3)];
            }
        }
        #pragma unroll
        for (int mh = 0; mh < 2; ++mh) {
            FragAB af;
            const int m   = wv * 32 + mh * 16 + (lane & 15);
            const int kh8 = (lane < 16) ? 0 : 8;
            #pragma unroll
            for (int j = 0; j < 8; ++j) {
                const int kk = ((j < 4) ? 0 : 16) + kh8 + 2 * (j & 3);
                af.u[j] = *(const unsigned int*)&As[m * AST + kk];
            }
            #pragma unroll
            for (int nt = 0; nt < 4; ++nt)
                acc[mh][nt] = wmma_bf16(af.v, bfr[nt].v, acc[mh][nt]);
        }
        __syncthreads();
    }

    // Epilogue: bias + store per output mode. C layout: lane%16 = N, VGPR r -> M=r(+8)
    const int rowhi = (lane < 16) ? 0 : 8;
    #pragma unroll
    for (int mh = 0; mh < 2; ++mh) {
        #pragma unroll
        for (int nt = 0; nt < 4; ++nt) {
            const int n = n0 + nt * 16 + (lane & 15);
            const float bv = bias[n];
            #pragma unroll
            for (int r = 0; r < 8; ++r) {
                const int m = m0 + wv * 32 + mh * 16 + r + rowhi;
                const float val = acc[mh][nt][r] + bv;
                if constexpr (MODE == 1) {
                    ((float*)Out)[(long)m * N + n] = val;
                } else {
                    const int b = m / SS, s = m % SS;
                    const int h = n / DHH, dh = n % DHH;
                    long idx;
                    if constexpr (MODE == 0)
                        idx = (((long)(b * HH + h) * SS + s) * DHH + dh);
                    else
                        idx = (((long)(b * HH + h) * DHH + dh) * SS + s);
                    ((__bf16*)Out)[idx] = (__bf16)val;
                }
            }
        }
    }
}

// ---------------------------------------------------------------------------
// Flash attention: per wave one 16-row query tile; stream 64-key tiles
// (16 WMMAs per iteration; softmax reductions / rescale amortized 2x vs 32-key).
// Q,K: [B,H,S,DH] bf16.  V: [B,H,DH,S] bf16 (transposed).
// Output written in [H,B,S,DH] flat order (reference's transpose(0,1).reshape bug)
// so the final GEMM can read it as a plain [B*S, D] row-major matrix.
// ---------------------------------------------------------------------------
__global__ __launch_bounds__(256)
void attn_kernel(const __bf16* __restrict__ Qh, const __bf16* __restrict__ Kh,
                 const __bf16* __restrict__ Vt, __bf16* __restrict__ Oh)
{
    constexpr int KT  = 64;                 // keys per iteration
    constexpr int PST = KT + 2;             // padded row stride (66)
    __shared__ __bf16 Ps[8 * 16 * PST];     // per-wave 16x64 P transpose scratch

    const int tid  = threadIdx.x;
    const int lane = tid & 31;
    const int wv   = tid >> 5;
    const int b    = blockIdx.y / HH;
    const int h    = blockIdx.y % HH;
    const int qs   = blockIdx.x * 128 + wv * 16;

    const __bf16* Q = Qh + (long)(b * HH + h) * SS * DHH;
    const __bf16* K = Kh + (long)(b * HH + h) * SS * DHH;
    const __bf16* V = Vt + (long)(b * HH + h) * DHH * SS;
    __bf16* P = &Ps[wv * 16 * PST];

    // Persistent Q A-fragments: 16 rows x 64 features = two 16x32 frags
    FragAB aq[2];
    {
        const int m   = qs + (lane & 15);
        const int kh8 = (lane < 16) ? 0 : 8;
        #pragma unroll
        for (int hf = 0; hf < 2; ++hf)
            #pragma unroll
            for (int j = 0; j < 8; ++j) {
                const int kk = hf * 32 + ((j < 4) ? 0 : 16) + kh8 + 2 * (j & 3);
                aq[hf].u[j] = *(const unsigned int*)&Q[m * DHH + kk];
            }
    }

    v8f acc[4];
    { v8f z = {}; acc[0]=z; acc[1]=z; acc[2]=z; acc[3]=z; }
    float mrow[8], lrow[8];
    #pragma unroll
    for (int r = 0; r < 8; ++r) { mrow[r] = -1e30f; lrow[r] = 0.0f; }

    for (int kt = 0; kt < SS; kt += KT) {
        // scores S = Q @ K^T : four 16x16 score frags (keys g*16..g*16+15)
        v8f sc[4];
        #pragma unroll
        for (int g = 0; g < 4; ++g) {
            v8f s = {};
            #pragma unroll
            for (int hf = 0; hf < 2; ++hf) {
                FragAB bk;
                const int key = kt + g * 16 + (lane & 15);
                const int kb  = (lane < 16) ? 0 : 16;
                #pragma unroll
                for (int j = 0; j < 8; ++j)
                    bk.u[j] = *(const unsigned int*)&K[key * DHH + hf * 32 + ((j < 4) ? 0 : 16) + (kb >> 1) + 2 * (j & 3)];
                s = wmma_bf16(aq[hf].v, bk.v, s);
            }
            sc[g] = s;
        }

        // Online softmax, per-row across the 16-lane half holding that row
        float escale[8];
        #pragma unroll
        for (int r = 0; r < 8; ++r) {
            float mx = fmaxf(fmaxf(sc[0][r], sc[1][r]), fmaxf(sc[2][r], sc[3][r]));
            #pragma unroll
            for (int msk = 8; msk >= 1; msk >>= 1)
                mx = fmaxf(mx, __shfl_xor(mx, msk, 16));
            const float mnew = fmaxf(mrow[r], mx);
            float sum = 0.0f;
            #pragma unroll
            for (int g = 0; g < 4; ++g) {
                const float p = __expf(sc[g][r] - mnew);
                sc[g][r] = p;
                sum += p;
            }
            #pragma unroll
            for (int msk = 8; msk >= 1; msk >>= 1)
                sum += __shfl_xor(sum, msk, 16);
            escale[r] = __expf(mrow[r] - mnew);
            lrow[r] = lrow[r] * escale[r] + sum;
            mrow[r] = mnew;
        }
        #pragma unroll
        for (int nt = 0; nt < 4; ++nt)
            #pragma unroll
            for (int r = 0; r < 8; ++r)
                acc[nt][r] *= escale[r];

        // Transpose P (C layout f32) -> A layout bf16 through per-wave LDS
        {
            const int col = lane & 15;
            const int rhi = (lane < 16) ? 0 : 8;
            #pragma unroll
            for (int g = 0; g < 4; ++g)
                #pragma unroll
                for (int r = 0; r < 8; ++r)
                    P[(r + rhi) * PST + g * 16 + col] = (__bf16)sc[g][r];
        }
        __asm__ volatile("s_wait_dscnt 0" ::: "memory");   // per-wave DS ordering
        FragAB ap[2];
        {
            const int m   = lane & 15;
            const int kh8 = (lane < 16) ? 0 : 8;
            #pragma unroll
            for (int hf = 0; hf < 2; ++hf)
                #pragma unroll
                for (int j = 0; j < 8; ++j) {
                    const int kk = hf * 32 + ((j < 4) ? 0 : 16) + kh8 + 2 * (j & 3);
                    ap[hf].u[j] = *(const unsigned int*)&P[m * PST + kk];
                }
        }

        // acc += P @ V  (V transposed: k-pairs contiguous per lane)
        #pragma unroll
        for (int nt = 0; nt < 4; ++nt) {
            const int dh = nt * 16 + (lane & 15);
            const int kb = (lane < 16) ? 0 : 16;
            #pragma unroll
            for (int hf = 0; hf < 2; ++hf) {
                FragAB bv;
                #pragma unroll
                for (int j = 0; j < 8; ++j)
                    bv.u[j] = *(const unsigned int*)&V[dh * SS + kt + hf * 32 + ((j < 4) ? 0 : 16) + (kb >> 1) + 2 * (j & 3)];
                acc[nt] = wmma_bf16(ap[hf].v, bv.v, acc[nt]);
            }
        }
    }

    // Normalize and store in [H,B,S,DH] flat order (bf16)
    {
        const int rhi = (lane < 16) ? 0 : 8;
        #pragma unroll
        for (int r = 0; r < 8; ++r) {
            const float inv = 1.0f / lrow[r];
            const int s = qs + r + rhi;
            const long base = (((long)h * BB + b) * SS + s) * DHH;
            #pragma unroll
            for (int nt = 0; nt < 4; ++nt)
                Oh[base + nt * 16 + (lane & 15)] = (__bf16)(acc[nt][r] * inv);
        }
    }
}

// ---------------------------------------------------------------------------
extern "C" void kernel_launch(void* const* d_in, const int* in_sizes, int n_in,
                              void* d_out, int out_size, void* d_ws, size_t ws_size,
                              hipStream_t stream)
{
    const float* q  = (const float*)d_in[0];
    const float* k  = (const float*)d_in[1];
    const float* v  = (const float*)d_in[2];
    const float* wq = (const float*)d_in[3];
    const float* bq = (const float*)d_in[4];
    const float* wk = (const float*)d_in[5];
    const float* bk = (const float*)d_in[6];
    const float* wv = (const float*)d_in[7];
    const float* bv = (const float*)d_in[8];
    const float* wo = (const float*)d_in[9];
    const float* bo = (const float*)d_in[10];

    const size_t HEAD_ELEMS = (size_t)BB * HH * SS * DHH;  // 4 Mi elements
    __bf16* Qh = (__bf16*)d_ws;
    __bf16* Kh = Qh + HEAD_ELEMS;
    __bf16* Vt = Kh + HEAD_ELEMS;
    __bf16* Ao = Vt + HEAD_ELEMS;

    const dim3 gg(MM / 256, DDIM / 64);
    gemm_kernel<0, true ><<<gg, 256, 0, stream>>>(q, wq, bq, Qh);   // Q head-split
    gemm_kernel<0, true ><<<gg, 256, 0, stream>>>(k, wk, bk, Kh);   // K head-split
    gemm_kernel<2, true ><<<gg, 256, 0, stream>>>(v, wv, bv, Vt);   // V transposed

    attn_kernel<<<dim3(SS / 128, BB * HH), 256, 0, stream>>>(Qh, Kh, Vt, Ao);

    gemm_kernel<1, false><<<gg, 256, 0, stream>>>(Ao, wo, bo, d_out); // out proj
}
